// MultiheadAttention_8005819039978
// MI455X (gfx1250) — matrix operand order repaired and checked by reading, not compiled, not run on hardware
//
#include <hip/hip_runtime.h>

#ifndef __has_builtin
#define __has_builtin(x) 0
#endif

#define B_     4
#define H_     16
#define N_     1024
#define DM_    1024
#define HD_    64
#define SCALE_ 0.125f           // 64^-0.5
#define NEG_INF_ (-1.0e9f)
#define LDSW   1028             // padded row stride (floats) to avoid LDS bank conflicts

typedef __attribute__((ext_vector_type(16))) __bf16 v16bf;
typedef __attribute__((ext_vector_type(8)))  __bf16 bf16x8;
typedef __attribute__((ext_vector_type(8)))  float  v8f;
typedef __attribute__((ext_vector_type(4)))  float  f32x4;

#if defined(__AMDGCN__) && __has_builtin(__builtin_amdgcn_tensor_load_to_lds) && \
    __has_builtin(__builtin_amdgcn_s_wait_tensorcnt)
#define HAVE_TDM 1
#else
#define HAVE_TDM 0
#endif

static __device__ inline v16bf join16(bf16x8 lo, bf16x8 hi) {
  return __builtin_shufflevector(lo, hi, 0,1,2,3,4,5,6,7,8,9,10,11,12,13,14,15);
}
static __device__ inline bf16x8 cvt8(f32x4 a, f32x4 b) {
  bf16x8 r;
  r[0]=(__bf16)a[0]; r[1]=(__bf16)a[1]; r[2]=(__bf16)a[2]; r[3]=(__bf16)a[3];
  r[4]=(__bf16)b[0]; r[5]=(__bf16)b[1]; r[6]=(__bf16)b[2]; r[7]=(__bf16)b[3];
  return r;
}
static __device__ inline v8f wmma_bf16(v16bf a, v16bf b, v8f c) {
  return __builtin_amdgcn_wmma_f32_16x16x32_bf16(false, a, false, b, (short)0, c, false, false);
}
// 16-bf16 fragment load, table order (two b128s at p and p+16)
static __device__ inline v16bf ldfrag(const __bf16* p) {
  return join16(*(const bf16x8*)(p), *(const bf16x8*)(p + 16));
}
// A fragment from an f32 row (convert to bf16)
static __device__ inline v16bf loadA_f32(const float* xr, int kb, int half) {
  f32x4 a0 = *(const f32x4*)(xr + kb + half*8);
  f32x4 a1 = *(const f32x4*)(xr + kb + half*8 + 4);
  f32x4 a2 = *(const f32x4*)(xr + kb + 16 + half*8);
  f32x4 a3 = *(const f32x4*)(xr + kb + 16 + half*8 + 4);
  return join16(cvt8(a0, a1), cvt8(a2, a3));
}

#if HAVE_TDM
typedef __attribute__((ext_vector_type(4))) unsigned int u32x4;
typedef __attribute__((ext_vector_type(4))) int          i32x4;
typedef __attribute__((ext_vector_type(8))) int          i32x8;
typedef __attribute__((address_space(3))) const char     lds_char_t;

static __device__ inline unsigned lds_addr_of(const void* p) {
  return (unsigned)(unsigned long long)(lds_char_t*)p;
}
// 2-D TDM: load a rows x row_elems tile of 2-byte elements into LDS.
static __device__ inline void tdm_load_tile_2d(unsigned lds_byte_addr, const void* gaddr,
                                               unsigned rows, unsigned row_elems,
                                               unsigned stride_elems, unsigned tensor_rows)
{
  unsigned long long ga = (unsigned long long)gaddr;
  u32x4 g0;
  g0[0] = 1u;                                                // count=1 (valid user D#)
  g0[1] = lds_byte_addr;                                     // lds_addr
  g0[2] = (unsigned)(ga & 0xffffffffu);                      // global_addr[31:0]
  g0[3] = (unsigned)((ga >> 32) & 0x01ffffffu) | (2u << 30); // addr[56:32] | type=2
  i32x8 g1 = {};
  g1[0] = (1 << 16);                                         // data_size=1 -> 2 bytes
  g1[1] = (int)((row_elems & 0xffffu) << 16);                // tensor_dim0 lo16
  g1[2] = (int)(((row_elems >> 16) & 0xffffu) |
                ((tensor_rows & 0xffffu) << 16));            // dim0 hi16 | tensor_dim1 lo16
  g1[3] = (int)(((tensor_rows >> 16) & 0xffffu) |
                ((row_elems & 0xffffu) << 16));              // dim1 hi16 | tile_dim0
  g1[4] = (int)(rows & 0xffffu);                             // tile_dim1 (tile_dim2=0)
  g1[5] = (int)stride_elems;                                 // tensor_dim0_stride lo32
  i32x4 gz = {};
#if __clang_major__ >= 23
  i32x8 gz8 = {};
  __builtin_amdgcn_tensor_load_to_lds(g0, g1, gz, gz, gz8, 0);
#else
  __builtin_amdgcn_tensor_load_to_lds(g0, g1, gz, gz, 0);
#endif
}
#endif // HAVE_TDM

// ---------------------------------------------------------------------------
// Kernel 0: one-shot weight conversion f32 -> bf16 (read-once/write-once, NT)
// ---------------------------------------------------------------------------
__global__ __launch_bounds__(256)
void wconv_kernel(const float* __restrict__ W, __bf16* __restrict__ Wbf)
{
  const size_t i = ((size_t)blockIdx.x * 256 + threadIdx.x) * 8;
  f32x4 a = __builtin_nontemporal_load((const f32x4*)(W + i));
  f32x4 b = __builtin_nontemporal_load((const f32x4*)(W + i + 4));
  bf16x8 r = cvt8(a, b);
  __builtin_nontemporal_store(r, (bf16x8*)(Wbf + i));
}

// ---------------------------------------------------------------------------
// Kernel 1: fused QKV projection, 2x-unrolled even/odd pipeline (no reg moves).
// Q,K -> [b][h][n][64];  V -> transposed [b][h][d][n]
// ---------------------------------------------------------------------------
__global__ __launch_bounds__(32)
void qkv_proj_kernel(const float* __restrict__ q, const float* __restrict__ k,
                     const float* __restrict__ v, const __bf16* __restrict__ Wbf,
                     const float* __restrict__ bias,
                     __bf16* __restrict__ qws, __bf16* __restrict__ kws,
                     __bf16* __restrict__ vTws)
{
  const int tid  = threadIdx.x;
  const int half = tid >> 4, ln = tid & 15;
  const int m0 = blockIdx.x * 16;
  const int o0 = blockIdx.y * 64;
  const int z  = blockIdx.z;
  const float* x = (z == 0) ? q : (z == 1) ? k : v;

  const float*  xr = x + (size_t)(m0 + ln) * DM_;
  const __bf16* wb = Wbf + (size_t)(z*DM_ + o0 + ln) * DM_ + half*8;
  v8f acc0 = {}, acc1 = {}, acc2 = {}, acc3 = {};

  // even stage (kb = 0)
  v16bf aE  = loadA_f32(xr, 0, half);
  v16bf bE0 = ldfrag(wb);
  v16bf bE1 = ldfrag(wb + (size_t)16*DM_);
  v16bf bE2 = ldfrag(wb + (size_t)32*DM_);
  v16bf bE3 = ldfrag(wb + (size_t)48*DM_);

  for (int kb = 0; kb < DM_; kb += 64) {
    // odd stage loads (kb+32) issue before even-stage WMMAs retire
    v16bf aO  = loadA_f32(xr, kb + 32, half);
    v16bf bO0 = ldfrag(wb + kb + 32);
    v16bf bO1 = ldfrag(wb + (size_t)16*DM_ + kb + 32);
    v16bf bO2 = ldfrag(wb + (size_t)32*DM_ + kb + 32);
    v16bf bO3 = ldfrag(wb + (size_t)48*DM_ + kb + 32);

    acc0 = wmma_bf16(aE, bE0, acc0);
    acc1 = wmma_bf16(aE, bE1, acc1);
    acc2 = wmma_bf16(aE, bE2, acc2);
    acc3 = wmma_bf16(aE, bE3, acc3);

    if (kb + 64 < DM_) {               // next even stage
      aE  = loadA_f32(xr, kb + 64, half);
      bE0 = ldfrag(wb + kb + 64);
      bE1 = ldfrag(wb + (size_t)16*DM_ + kb + 64);
      bE2 = ldfrag(wb + (size_t)32*DM_ + kb + 64);
      bE3 = ldfrag(wb + (size_t)48*DM_ + kb + 64);
    }
    acc0 = wmma_bf16(aO, bO0, acc0);
    acc1 = wmma_bf16(aO, bO1, acc1);
    acc2 = wmma_bf16(aO, bO2, acc2);
    acc3 = wmma_bf16(aO, bO3, acc3);
  }

  v8f accs[4] = {acc0, acc1, acc2, acc3};
  #pragma unroll
  for (int nt = 0; nt < 4; ++nt) {
    const int o  = o0 + nt*16 + ln;
    const float bv = bias[z*DM_ + o];
    const int h = o >> 6, d = o & 63;
    #pragma unroll
    for (int r = 0; r < 8; ++r) {
      const int m  = m0 + r + 8*half;
      const int bb = m >> 10, n = m & 1023;
      const __bf16 val = (__bf16)(accs[nt][r] + bv);
      if (z == 2)       // V transposed: [b][h][d][n]
        vTws[(((size_t)bb*H_ + h)*HD_ + d)*N_ + n] = val;
      else if (z == 0)
        qws[(((size_t)bb*H_ + h)*N_ + n)*HD_ + d] = val;
      else
        kws[(((size_t)bb*H_ + h)*N_ + n)*HD_ + d] = val;
    }
  }
}

// ---------------------------------------------------------------------------
// Kernel 2: fused attention for one (b, h, 16-query tile).
// K tiles staged to LDS by the Tensor Data Mover (double-buffered, TENSORcnt).
// ---------------------------------------------------------------------------
__global__ __launch_bounds__(32)
void attn_kernel(const __bf16* __restrict__ qws, const __bf16* __restrict__ kws,
                 const __bf16* __restrict__ vTws,
                 const unsigned char* __restrict__ mask,
                 const float* __restrict__ gdists,
                 const float* __restrict__ w_graph,
                 float* __restrict__ att_out, __bf16* __restrict__ yws)
{
  extern __shared__ float S[];                 // [16][LDSW] f32 score rows
  __shared__ __bf16 KT[2][16 * HD_];           // TDM-staged K tiles (2 x 2KB)
  const int tid  = threadIdx.x;
  const int half = tid >> 4, ln = tid & 15;
  const int n0 = blockIdx.x * 16;
  const int h  = blockIdx.y;
  const int b  = blockIdx.z;
  const size_t bh = (size_t)b*H_ + h;

  f32x4 g0 = *(const f32x4*)(w_graph + h*8);
  f32x4 g1 = *(const f32x4*)(w_graph + h*8 + 4);

  // Q fragments (K-dim = head_dim = 64 -> two 16x32 A fragments)
  const __bf16* qrow = qws + (bh*N_ + n0 + ln) * HD_;
  v16bf a0 = join16(*(const bf16x8*)(qrow      + half*16),
                    *(const bf16x8*)(qrow      + half*16 + 8));
  v16bf a1 = join16(*(const bf16x8*)(qrow + 32 + half*16),
                    *(const bf16x8*)(qrow + 32 + half*16 + 8));

  bool rowMasked[8];
  #pragma unroll
  for (int r = 0; r < 8; ++r)
    rowMasked[r] = mask[b*N_ + n0 + r + 8*half] != 0;

  const __bf16* kpanel = kws + bh*N_*HD_;
#if HAVE_TDM
  tdm_load_tile_2d(lds_addr_of(&KT[0][0]), kpanel, 16, HD_, HD_, N_);
#endif

  // ---- phase 1: scores into LDS ----
  for (int mt = 0; mt < N_/16; ++mt) {
    const int m0x = mt*16;
#if HAVE_TDM
    if (mt + 1 < N_/16) {
      tdm_load_tile_2d(lds_addr_of(&KT[(mt+1)&1][0]),
                       kpanel + (size_t)(m0x + 16)*HD_, 16, HD_, HD_, N_ - m0x - 16);
      __builtin_amdgcn_s_wait_tensorcnt((short)1);   // tile mt ready (in-order TDM)
    } else {
      __builtin_amdgcn_s_wait_tensorcnt((short)0);
    }
    const __bf16* krow = &KT[mt & 1][ln * HD_];
#else
    const __bf16* krow = kpanel + (size_t)(m0x + ln) * HD_;
#endif
    v16bf b0 = join16(*(const bf16x8*)(krow      + half*16),
                      *(const bf16x8*)(krow      + half*16 + 8));
    v16bf b1 = join16(*(const bf16x8*)(krow + 32 + half*16),
                      *(const bf16x8*)(krow + 32 + half*16 + 8));
    v8f c = {};
    c = wmma_bf16(a0, b0, c);
    c = wmma_bf16(a1, b1, c);

    const int mcol = m0x + ln;
    // stream-ahead hint for the read-once gdists
    __builtin_prefetch(gdists + (((size_t)b*N_ + n0 + 8*half)*N_ + mcol + 16)*8, 0, 0);
    #pragma unroll
    for (int r = 0; r < 8; ++r) {
      const int M  = r + 8*half;
      const int nq = n0 + M;
      const float* gp = gdists + (((size_t)b*N_ + nq)*N_ + mcol)*8;
      f32x4 d0 = __builtin_nontemporal_load((const f32x4*)gp);
      f32x4 d1 = __builtin_nontemporal_load((const f32x4*)(gp + 4));
      float gbias = d0[0]*g0[0] + d0[1]*g0[1] + d0[2]*g0[2] + d0[3]*g0[3]
                  + d1[0]*g1[0] + d1[1]*g1[1] + d1[2]*g1[2] + d1[3]*g1[3];
      float sv = c[r]*SCALE_ + gbias;
      if (rowMasked[r]) sv = NEG_INF_;
      S[M*LDSW + mcol] = sv;
    }
  }

  // ---- phase 2: softmax per row; att out (NT stores); P kept in LDS ----
  float* attRow = att_out + (bh*N_ + n0) * N_;
  for (int row = 0; row < 16; ++row) {
    float* Sr = S + row*LDSW;
    float mx = -3.4e38f;
    for (int i = tid; i < N_; i += 32) mx = fmaxf(mx, Sr[i]);
    #pragma unroll
    for (int off = 16; off > 0; off >>= 1) mx = fmaxf(mx, __shfl_xor(mx, off, 32));
    float sum = 0.f;
    for (int i = tid; i < N_; i += 32) {
      float e = __expf(Sr[i] - mx);
      Sr[i] = e;
      sum += e;
    }
    #pragma unroll
    for (int off = 16; off > 0; off >>= 1) sum += __shfl_xor(sum, off, 32);
    const float inv = 1.0f / sum;
    float* arow = attRow + (size_t)row * N_;
    for (int i = tid*4; i < N_; i += 128) {
      f32x4 p = *(f32x4*)(Sr + i) * inv;
      *(f32x4*)(Sr + i) = p;                              // P for PV
      __builtin_nontemporal_store(p, (f32x4*)(arow + i)); // att (write-once)
    }
  }

  // ---- phase 3: O = P @ V, V read transposed (contiguous B fragments) ----
  v8f o0a = {}, o1a = {}, o2a = {}, o3a = {};
  for (int kt = 0; kt < N_; kt += 32) {
    const float* pr = S + ln*LDSW + kt;
    f32x4 p0 = *(const f32x4*)(pr + half*16);
    f32x4 p1 = *(const f32x4*)(pr + half*16 + 4);
    f32x4 p2 = *(const f32x4*)(pr + half*16 + 8);
    f32x4 p3 = *(const f32x4*)(pr + half*16 + 12);
    v16bf ap = join16(cvt8(p0, p1), cvt8(p2, p3));
    const __bf16* vrow = vTws + (bh*HD_ + ln)*N_ + kt + half*16;
    o0a = wmma_bf16(ap, join16(*(const bf16x8*)(vrow),
                               *(const bf16x8*)(vrow + 8)), o0a);
    o1a = wmma_bf16(ap, join16(*(const bf16x8*)(vrow + (size_t)16*N_),
                               *(const bf16x8*)(vrow + (size_t)16*N_ + 8)), o1a);
    o2a = wmma_bf16(ap, join16(*(const bf16x8*)(vrow + (size_t)32*N_),
                               *(const bf16x8*)(vrow + (size_t)32*N_ + 8)), o2a);
    o3a = wmma_bf16(ap, join16(*(const bf16x8*)(vrow + (size_t)48*N_),
                               *(const bf16x8*)(vrow + (size_t)48*N_ + 8)), o3a);
  }
  v8f os[4] = {o0a, o1a, o2a, o3a};
  #pragma unroll
  for (int nt = 0; nt < 4; ++nt) {
    #pragma unroll
    for (int r = 0; r < 8; ++r) {
      const int M = r + 8*half;
      yws[((size_t)b*N_ + n0 + M)*DM_ + h*HD_ + nt*16 + ln] = (__bf16)os[nt][r];
    }
  }
}

// ---------------------------------------------------------------------------
// Kernel 3: output projection, 2x-unrolled even/odd pipeline.
// ---------------------------------------------------------------------------
__global__ __launch_bounds__(32)
void out_proj_kernel(const __bf16* __restrict__ yws, const __bf16* __restrict__ Wbf,
                     const float* __restrict__ bias, float* __restrict__ out)
{
  const int tid  = threadIdx.x;
  const int half = tid >> 4, ln = tid & 15;
  const int m0 = blockIdx.x * 16;
  const int o0 = blockIdx.y * 64;
  const __bf16* yr = yws + (size_t)(m0 + ln) * DM_;
  const __bf16* wb = Wbf + (size_t)(3*DM_ + o0 + ln) * DM_ + half*8;
  v8f acc0 = {}, acc1 = {}, acc2 = {}, acc3 = {};

  v16bf aE  = join16(*(const bf16x8*)(yr + half*8),
                     *(const bf16x8*)(yr + 16 + half*8));
  v16bf bE0 = ldfrag(wb);
  v16bf bE1 = ldfrag(wb + (size_t)16*DM_);
  v16bf bE2 = ldfrag(wb + (size_t)32*DM_);
  v16bf bE3 = ldfrag(wb + (size_t)48*DM_);

  for (int kb = 0; kb < DM_; kb += 64) {
    v16bf aO  = join16(*(const bf16x8*)(yr + kb + 32 + half*8),
                       *(const bf16x8*)(yr + kb + 48 + half*8));
    v16bf bO0 = ldfrag(wb + kb + 32);
    v16bf bO1 = ldfrag(wb + (size_t)16*DM_ + kb + 32);
    v16bf bO2 = ldfrag(wb + (size_t)32*DM_ + kb + 32);
    v16bf bO3 = ldfrag(wb + (size_t)48*DM_ + kb + 32);

    acc0 = wmma_bf16(aE, bE0, acc0);
    acc1 = wmma_bf16(aE, bE1, acc1);
    acc2 = wmma_bf16(aE, bE2, acc2);
    acc3 = wmma_bf16(aE, bE3, acc3);

    if (kb + 64 < DM_) {
      aE  = join16(*(const bf16x8*)(yr + kb + 64 + half*8),
                   *(const bf16x8*)(yr + kb + 80 + half*8));
      bE0 = ldfrag(wb + kb + 64);
      bE1 = ldfrag(wb + (size_t)16*DM_ + kb + 64);
      bE2 = ldfrag(wb + (size_t)32*DM_ + kb + 64);
      bE3 = ldfrag(wb + (size_t)48*DM_ + kb + 64);
    }
    acc0 = wmma_bf16(aO, bO0, acc0);
    acc1 = wmma_bf16(aO, bO1, acc1);
    acc2 = wmma_bf16(aO, bO2, acc2);
    acc3 = wmma_bf16(aO, bO3, acc3);
  }

  v8f accs[4] = {acc0, acc1, acc2, acc3};
  #pragma unroll
  for (int nt = 0; nt < 4; ++nt) {
    const int o = o0 + nt*16 + ln;
    const float bv = bias[3*DM_ + o];
    #pragma unroll
    for (int r = 0; r < 8; ++r) {
      const int m = m0 + r + 8*half;
      __builtin_nontemporal_store(accs[nt][r] + bv, out + (size_t)m*DM_ + o);
    }
  }
}

// ---------------------------------------------------------------------------
extern "C" void kernel_launch(void* const* d_in, const int* in_sizes, int n_in,
                              void* d_out, int out_size, void* d_ws, size_t ws_size,
                              hipStream_t stream)
{
  (void)in_sizes; (void)n_in; (void)out_size; (void)ws_size;
  const float*         q    = (const float*)d_in[0];
  const float*         k    = (const float*)d_in[1];
  const float*         v    = (const float*)d_in[2];
  const unsigned char* mask = (const unsigned char*)d_in[3];   // jax bool -> 1 byte
  const float*         gd   = (const float*)d_in[4];
  const float*         W    = (const float*)d_in[5];
  const float*         bias = (const float*)d_in[6];
  const float*         wg   = (const float*)d_in[7];

  char* ws = (char*)d_ws;
  const size_t proj_bytes = (size_t)B_ * N_ * DM_ * sizeof(__bf16);   // 8 MB each
  __bf16* qws  = (__bf16*)(ws);
  __bf16* kws  = (__bf16*)(ws + proj_bytes);
  __bf16* vTws = (__bf16*)(ws + 2*proj_bytes);
  __bf16* yws  = (__bf16*)(ws + 3*proj_bytes);
  __bf16* Wbf  = (__bf16*)(ws + 4*proj_bytes);                        // 4096x1024 bf16

  float* out = (float*)d_out;                         // [B,N,DM] f32
  float* att = out + (size_t)B_ * N_ * DM_;           // [B,H,N,N] f32

  // 0) weights -> bf16 (4M elements, 8 per thread)
  wconv_kernel<<<dim3((4*DM_*DM_)/(256*8)), dim3(256), 0, stream>>>(W, Wbf);

  dim3 blk(32, 1, 1);
  // 1) QKV projections
  qkv_proj_kernel<<<dim3((B_*N_)/16, DM_/64, 3), blk, 0, stream>>>(
      q, k, v, Wbf, bias, qws, kws, vTws);

  // 2) fused attention (scores+bias+mask+softmax+att+PV)
  const int ldsBytes = 16 * LDSW * (int)sizeof(float);   // ~64.3 KB dynamic
  (void)hipFuncSetAttribute((const void*)attn_kernel,
                            hipFuncAttributeMaxDynamicSharedMemorySize, ldsBytes);
  attn_kernel<<<dim3(N_/16, H_, B_), blk, ldsBytes, stream>>>(
      qws, kws, vTws, mask, gd, wg, att, yws);

  // 3) output projection
  out_proj_kernel<<<dim3((B_*N_)/16, DM_/64), blk, 0, stream>>>(
      yws, Wbf, bias, out);
}
